// META1_86397562127205
// MI455X (gfx1250) — matrix operand order repaired and checked by reading, not compiled, not run on hardware
//
#include <hip/hip_runtime.h>
#include <hip/hip_bf16.h>

typedef __attribute__((ext_vector_type(16))) _Float16 v16h;
typedef __attribute__((ext_vector_type(8)))  float    v8f;

__device__ __forceinline__ void wave_lds_fence() {
  __builtin_amdgcn_wave_barrier();
  asm volatile("s_wait_dscnt 0" ::: "memory");
  __builtin_amdgcn_wave_barrier();
}

// single-instruction ReLU: med3(x, 0, +FLT_MAX)
__device__ __forceinline__ float relu(float v) {
  return __builtin_amdgcn_fmed3f(v, 0.f, 3.402823466e+38f);
}

// Staging permutation: swap bits 3 and 4 of K (within a 32-wide K block).
// With this storage order, A-fragment element e of lane hi sits at
// base + hi*16 + e, i.e. each lane's 16 halves are contiguous (32 B).
__device__ __forceinline__ int swap34(int k) {
  return (k & ~24) | ((k & 8) << 1) | ((k & 16) >> 1);
}

__device__ __forceinline__ v8f wmma_f16(v16h a, v16h b, v8f c) {
  // D = A(16x32 f16) * B(32x16 f16) + C(16x16 f32)
  return __builtin_amdgcn_wmma_f32_16x16x32_f16(
      /*neg_a=*/false, a, /*neg_b=*/false, b,
      /*c_mod=*/(short)0, c, /*reuse_a=*/false, /*reuse_b=*/false);
}

// B fragment (32x16 f16, K x N): lane = hi*16 + n, element e holds K = hi*8 + (e&7) + (e>>3)*16
__device__ __forceinline__ v16h load_bfrag(const _Float16* W, int npad,
                                           int ktile, int n, int hi) {
  v16h b;
  const int kbase = ktile * 32 + hi * 8;
#pragma unroll
  for (int e = 0; e < 16; ++e) b[e] = W[(kbase + (e & 7) + ((e >> 3) << 4)) * npad + n];
  return b;
}

// A fragment from swap34-permuted staging: one contiguous 32-byte load.
__device__ __forceinline__ v16h load_afrag(const _Float16* S, int rowstride,
                                           int m, int hi, int kbase) {
  return *(const v16h*)(S + m * rowstride + kbase + hi * 16);
}

// Biases are folded into the GEMMs as an augmented K row:
//   edge L1: W[9][n]  = eb1[n], A(K=9)  = 1
//   node L1: W[18][n] = nb1[n], A(K=18) = 1 (node staging, static)
//   both L2: W[50][n] = bias,   A(K=50) = 1 (layer-1 staging col 50 = 1)
__global__ __launch_bounds__(256) void fused_edge_node(
    const float* __restrict__ x, const float* __restrict__ edge_attr,
    const float* __restrict__ ew1, const float* __restrict__ eb1,
    const float* __restrict__ ew2, const float* __restrict__ eb2,
    const float* __restrict__ nw1, const float* __restrict__ nb1,
    const float* __restrict__ nw2, const float* __restrict__ nb2,
    const int* __restrict__ eidx,
    float* __restrict__ x2_sum, float* __restrict__ x2_cnt, int n_edges) {
  __shared__ _Float16 sEW1[32 * 64];      // 9x50  (+bias row 9)  -> 32x64
  __shared__ _Float16 sEW2[64 * 16];      // 50x15 (+bias row 50) -> 64x16
  __shared__ _Float16 sNW1[32 * 64];      // 18x50 (+bias row 18) -> 32x64
  __shared__ _Float16 sNW2[64 * 16];      // 50x15 (+bias row 50) -> 64x16
  __shared__ _Float16 sStage1[8][16 * 64];  // per-wave layer-1 out (swap34)
  __shared__ _Float16 sStageN[8][16 * 32];  // per-wave node input (swap34)

  const int tid = threadIdx.x;
  for (int i = tid; i < 32 * 64; i += 256) {
    const int k = i >> 6, n = i & 63;
    float ve = 0.f, vn = 0.f;
    if (n < 50) {
      if (k < 9)  ve = ew1[k * 50 + n];
      if (k == 9) ve = eb1[n];
      if (k < 18)  vn = nw1[k * 50 + n];
      if (k == 18) vn = nb1[n];
    }
    sEW1[i] = (_Float16)ve;
    sNW1[i] = (_Float16)vn;
  }
  for (int i = tid; i < 64 * 16; i += 256) {
    const int k = i >> 4, n = i & 15;
    float ve = 0.f, vn = 0.f;
    if (n < 15) {
      if (k < 50)  { ve = ew2[k * 15 + n]; vn = nw2[k * 15 + n]; }
      if (k == 50) { ve = eb2[n];          vn = nb2[n]; }
    }
    sEW2[i] = (_Float16)ve;
    sNW2[i] = (_Float16)vn;
  }
  __syncthreads();

  const int lane = tid & 31;
  const int wv = tid >> 5;
  const int nl = lane & 15;   // N column (or M row for A loads)
  const int hi = lane >> 4;
  _Float16* S1 = &sStage1[wv][0];
  _Float16* SN = &sStageN[wv][0];
  const _Float16 one = (_Float16)1.f;

  // Static part of node-input staging: K=18 -> 1.0 (bias row), K=19..31 -> 0.
  // Stored at swap34 positions; never overwritten inside the tile loop.
  for (int i = lane; i < 16 * 14; i += 32) {
    const int row = i / 14, K = 18 + i % 14;
    SN[row * 32 + swap34(K)] = (K == 18) ? one : (_Float16)0.f;
  }

  // Loop-invariant B fragments (held in VGPRs for the whole kernel)
  v16h bE1[4], bN1[4], bE2[2], bN2[2];
#pragma unroll
  for (int t = 0; t < 4; ++t) {
    bE1[t] = load_bfrag(sEW1, 64, 0, t * 16 + nl, hi);
    bN1[t] = load_bfrag(sNW1, 64, 0, t * 16 + nl, hi);
  }
#pragma unroll
  for (int kk = 0; kk < 2; ++kk) {
    bE2[kk] = load_bfrag(sEW2, 16, kk, nl, hi);
    bN2[kk] = load_bfrag(sNW2, 16, kk, nl, hi);
  }
  wave_lds_fence();

  const int ntiles = n_edges >> 4;              // 16 edges per tile
  const int gwave = blockIdx.x * 8 + wv;
  const int gwaves = gridDim.x * 8;

  for (int t = gwave; t < ntiles; t += gwaves) {
    const long base = (long)t << 4;
    if (t + gwaves < ntiles)
      __builtin_prefetch(&edge_attr[(((long)(t + gwaves)) << 4) * 3], 0, 1);

    const long e_my = base + nl;
    const int r_my = eidx[e_my];
    const int c_my = eidx[n_edges + e_my];

    // ---- edge MLP layer 1: A = [x[row] x[col] edge_attr 1] (K=10 pad 32) --
    // hi=0 lane elements 0..7 are K=0..7; hi=1 element 0 is K=8, element 1 K=9.
    v16h a1 = {};
    if (hi == 0) {
      a1[0] = (_Float16)x[r_my * 3 + 0];
      a1[1] = (_Float16)x[r_my * 3 + 1];
      a1[2] = (_Float16)x[r_my * 3 + 2];
      a1[3] = (_Float16)x[c_my * 3 + 0];
      a1[4] = (_Float16)x[c_my * 3 + 1];
      a1[5] = (_Float16)x[c_my * 3 + 2];
      a1[6] = (_Float16)edge_attr[e_my * 3 + 0];
      a1[7] = (_Float16)edge_attr[e_my * 3 + 1];
    } else {
      a1[0] = (_Float16)edge_attr[e_my * 3 + 2];
      a1[1] = one;                       // bias row K=9
    }

#pragma unroll
    for (int nt = 0; nt < 4; ++nt) {      // N = 64 (50 valid + bias already in)
      v8f acc = {};
      acc = wmma_f16(a1, bE1[nt], acc);
      const int n = nt * 16 + nl;
      const int pos = (n & 32) | swap34(n & 31);
      const bool biascol = (n == 50);     // feed 1.0 into layer-2 bias row
#pragma unroll
      for (int r = 0; r < 8; ++r) {       // D: elem r -> (M=r+hi*8, N=n)
        const int M = r + hi * 8;
        S1[M * 64 + pos] = biascol ? one : (_Float16)relu(acc[r]);
      }
    }
    wave_lds_fence();

    // ---- edge MLP layer 2: K=51 pad 64 -> edge_attr2 (bias included) ----
    v16h a2lo = load_afrag(S1, 64, nl, hi, 0);
    v16h a2hi = load_afrag(S1, 64, nl, hi, 32);
    wave_lds_fence();
    v8f e2 = {};
    e2 = wmma_f16(a2lo, bE2[0], e2);
    e2 = wmma_f16(a2hi, bE2[1], e2);

    // ---- stage node input: K 0-2 = x[col], K 3-17 = edge_attr2 ----
    if (hi == 0) {                         // swap34(0..2) == 0..2
      SN[nl * 32 + 0] = (_Float16)x[c_my * 3 + 0];
      SN[nl * 32 + 1] = (_Float16)x[c_my * 3 + 1];
      SN[nl * 32 + 2] = (_Float16)x[c_my * 3 + 2];
    }
    if (nl < 15) {
      const int posE = swap34(3 + nl);
#pragma unroll
      for (int r = 0; r < 8; ++r) {
        const int M = r + hi * 8;
        SN[M * 32 + posE] = (_Float16)e2[r];
      }
    }
    wave_lds_fence();

    // ---- node MLP layer 1: K=19 pad 32, N=64 ----
    v16h a3 = load_afrag(SN, 32, nl, hi, 0);
    wave_lds_fence();
#pragma unroll
    for (int nt = 0; nt < 4; ++nt) {
      v8f acc = {};
      acc = wmma_f16(a3, bN1[nt], acc);
      const int n = nt * 16 + nl;
      const int pos = (n & 32) | swap34(n & 31);
      const bool biascol = (n == 50);
#pragma unroll
      for (int r = 0; r < 8; ++r) {
        const int M = r + hi * 8;
        S1[M * 64 + pos] = biascol ? one : (_Float16)relu(acc[r]);
      }
    }
    wave_lds_fence();

    // ---- node MLP layer 2: K=51 pad 64 -> msg (bias included) ----
    v16h a4lo = load_afrag(S1, 64, nl, hi, 0);
    v16h a4hi = load_afrag(S1, 64, nl, hi, 32);
    wave_lds_fence();
    v8f msg = {};
    msg = wmma_f16(a4lo, bN2[0], msg);
    msg = wmma_f16(a4hi, bN2[1], msg);

    // ---- scatter-add msg over row[] + per-node edge count ----
#pragma unroll
    for (int r = 0; r < 8; ++r) {
      const int M = r + hi * 8;
      const int ro = eidx[base + M];
      if (nl < 15) {
        atomicAdd(&x2_sum[ro * 16 + nl], msg[r]);
      } else {
        atomicAdd(&x2_cnt[ro], 1.0f);
      }
    }
  }
}

__global__ void zero_kernel(float* __restrict__ p, int n) {
  int i = blockIdx.x * blockDim.x + threadIdx.x;
  const int stride = gridDim.x * blockDim.x;
  for (; i < n; i += stride) p[i] = 0.f;
}

__global__ void node_pool(const float* __restrict__ x2_sum,
                          const float* __restrict__ x2_cnt,
                          const int* __restrict__ batch,
                          float* __restrict__ u_sum, float* __restrict__ u_cnt,
                          int n_nodes) {
  const int i = blockIdx.x * blockDim.x + threadIdx.x;
  if (i >= n_nodes) return;
  const float inv = 1.f / fmaxf(x2_cnt[i], 1.f);
  const int g = batch[i];
#pragma unroll
  for (int f = 0; f < 15; ++f)
    atomicAdd(&u_sum[g * 16 + f], x2_sum[i * 16 + f] * inv);
  atomicAdd(&u_cnt[g], 1.f);
}

__global__ __launch_bounds__(512) void head_kernel(
    const float* __restrict__ u_sum, const float* __restrict__ u_cnt,
    const float* __restrict__ fc1_w, const float* __restrict__ fc1_b,
    const float* __restrict__ gamma, const float* __restrict__ beta,
    const float* __restrict__ fc2_w, const float* __restrict__ fc2_b,
    float* __restrict__ out, int num_graphs) {
  __shared__ float sS[10], sQ[10];
  const int g = threadIdx.x;  // one thread per graph, blockDim == num_graphs
  if (g < 10) { sS[g] = 0.f; sQ[g] = 0.f; }
  __syncthreads();

  const float inv = 1.f / fmaxf(u_cnt[g], 1.f);
  float u[15];
#pragma unroll
  for (int f = 0; f < 15; ++f) u[f] = u_sum[g * 16 + f] * inv;

  float h[10];
#pragma unroll
  for (int j = 0; j < 10; ++j) {
    float a = fc1_b[j];
#pragma unroll
    for (int f = 0; f < 15; ++f) a += u[f] * fc1_w[f * 10 + j];
    h[j] = a;
    atomicAdd(&sS[j], a);
    atomicAdd(&sQ[j], a * a);
  }
  __syncthreads();

  const float invG = 1.f / (float)num_graphs;
#pragma unroll
  for (int j = 0; j < 10; ++j) {
    const float mu = sS[j] * invG;
    const float var = sQ[j] * invG - mu * mu;
    const float v = gamma[j] * (h[j] - mu) * rsqrtf(var + 1e-5f) + beta[j];
    h[j] = fmaxf(v, 0.f);
  }

  float lg[6];
  float m = -1e30f;
#pragma unroll
  for (int c = 0; c < 6; ++c) {
    float a = fc2_b[c];
#pragma unroll
    for (int j = 0; j < 10; ++j) a += h[j] * fc2_w[j * 6 + c];
    lg[c] = a;
    m = fmaxf(m, a);
  }
  float s = 0.f;
#pragma unroll
  for (int c = 0; c < 6; ++c) s += __expf(lg[c] - m);
  const float lse = m + __logf(s);
#pragma unroll
  for (int c = 0; c < 6; ++c) out[g * 6 + c] = lg[c] - lse;
}

extern "C" void kernel_launch(void* const* d_in, const int* in_sizes, int n_in,
                              void* d_out, int out_size, void* d_ws,
                              size_t ws_size, hipStream_t stream) {
  const float* x = (const float*)d_in[0];
  const float* edge_attr = (const float*)d_in[1];
  const float* ew1 = (const float*)d_in[2];
  const float* eb1 = (const float*)d_in[3];
  const float* ew2 = (const float*)d_in[4];
  const float* eb2 = (const float*)d_in[5];
  const float* nw1 = (const float*)d_in[6];
  const float* nb1 = (const float*)d_in[7];
  const float* nw2 = (const float*)d_in[8];
  const float* nb2 = (const float*)d_in[9];
  const float* fc1_w = (const float*)d_in[10];
  const float* fc1_b = (const float*)d_in[11];
  const float* bn_g = (const float*)d_in[12];
  const float* bn_b = (const float*)d_in[13];
  const float* fc2_w = (const float*)d_in[14];
  const float* fc2_b = (const float*)d_in[15];
  const int* eidx = (const int*)d_in[16];
  const int* batch = (const int*)d_in[17];

  const int n_nodes = in_sizes[0] / 3;
  const int n_edges = in_sizes[16] / 2;
  const int num_graphs = out_size / 6;  // 512

  // workspace layout (floats): x2_sum[N*16] | x2_cnt[N] | u_sum[G*16] | u_cnt[G]
  float* x2_sum = (float*)d_ws;
  float* x2_cnt = x2_sum + (size_t)n_nodes * 16;
  float* u_sum = x2_cnt + n_nodes;
  float* u_cnt = u_sum + (size_t)num_graphs * 16;
  const int zero_n = n_nodes * 16 + n_nodes + num_graphs * 16 + num_graphs;

  zero_kernel<<<512, 256, 0, stream>>>(x2_sum, zero_n);
  fused_edge_node<<<1024, 256, 0, stream>>>(x, edge_attr, ew1, eb1, ew2, eb2,
                                            nw1, nb1, nw2, nb2, eidx, x2_sum,
                                            x2_cnt, n_edges);
  node_pool<<<(n_nodes + 255) / 256, 256, 0, stream>>>(x2_sum, x2_cnt, batch,
                                                       u_sum, u_cnt, n_nodes);
  head_kernel<<<1, num_graphs, 0, stream>>>(u_sum, u_cnt, fc1_w, fc1_b, bn_g,
                                            bn_b, fc2_w, fc2_b, (float*)d_out,
                                            num_graphs);
}
